// StackLSTM_73942156968529
// MI455X (gfx1250) — matrix operand order repaired and checked by reading, not compiled, not run on hardware
//
#include <hip/hip_runtime.h>
#include <hip/hip_bf16.h>

typedef __bf16 bf16_t;
typedef __attribute__((ext_vector_type(16))) __bf16 v16bf;
typedef __attribute__((ext_vector_type(8)))  float  v8f;

#define VOCAB 5000
#define EMB   128
#define HID   512
#define STK   128
#define BSZ   128
#define TLEN  128
#define G4    2048   // 4*HID
#define KZ    768    // EMB + STK + HID
#define NWG   64
#define NTHR  256
#define BCP   776    // padded LDS row (bf16) for Wc slab  (768+8 -> bank-staggered)
#define BVP   520    // padded LDS row (bf16) for Wval slab (512+8)

// ---------------------------------------------------------------------------
// helpers
// ---------------------------------------------------------------------------
__device__ __forceinline__ float sigf(float x) { return 1.0f / (1.0f + __expf(-x)); }

// A fragment (16x32 bf16, M x K) from row-major [.., ld] bf16 matrix.
__device__ __forceinline__ v16bf load_a16(const bf16_t* __restrict__ P, int ld,
                                          int m0, int k0, int lane) {
  const bf16_t* p = P + (long)(m0 + (lane & 15)) * ld + k0 + ((lane >> 4) << 3);
  v16bf a;
#pragma unroll
  for (int i = 0; i < 8; ++i) a[i] = p[i];
#pragma unroll
  for (int i = 0; i < 8; ++i) a[8 + i] = p[16 + i];
  return a;
}

// B fragment (32x16 bf16, K x N) from row-major [N, K] storage (global or LDS).
__device__ __forceinline__ v16bf load_b16(const bf16_t* P, int ld,
                                          int n0, int k0, int lane) {
  const bf16_t* p = P + (long)(n0 + (lane & 15)) * ld + k0 + ((lane >> 4) << 3);
  v16bf b;
#pragma unroll
  for (int i = 0; i < 8; ++i) b[i] = p[i];
#pragma unroll
  for (int i = 0; i < 8; ++i) b[8 + i] = p[16 + i];
  return b;
}

__device__ __forceinline__ void grid_barrier(unsigned* bar, unsigned nwg) {
  __syncthreads();
  if (threadIdx.x == 0) {
    __threadfence();
    volatile unsigned* vgen = bar + 1;
    unsigned g = *vgen;
    if (atomicAdd(bar, 1u) == nwg - 1u) {
      *((volatile unsigned*)bar) = 0u;
      __threadfence();
      atomicAdd(bar + 1, 1u);
    } else {
      while (*vgen == g) { __builtin_amdgcn_s_sleep(2); }
    }
    __threadfence();
  }
  __syncthreads();
}

// async 16B global -> LDS copy (CDNA5 ASYNCcnt path)
__device__ __forceinline__ void async_copy_b128(unsigned lds_addr, const void* gptr) {
  asm volatile("global_load_async_to_lds_b128 %0, %1, off"
               :: "v"(lds_addr), "v"(gptr) : "memory");
}
__device__ __forceinline__ void wait_async0() {
  asm volatile("s_wait_asynccnt 0" ::: "memory");
}

// ---------------------------------------------------------------------------
// init: bf16 weight packing, embedding gather, state reset (every call)
// ---------------------------------------------------------------------------
__global__ void init_kernel(const int* __restrict__ x, const float* __restrict__ emb,
                            const float* __restrict__ W_ih, const float* __restrict__ b_ih,
                            const float* __restrict__ W_hh, const float* __restrict__ b_hh,
                            const float* __restrict__ W_val_f,
                            bf16_t* __restrict__ Wc, float* __restrict__ bc,
                            bf16_t* __restrict__ Wval, bf16_t* __restrict__ exbf,
                            bf16_t* __restrict__ Z, float* __restrict__ h,
                            float* __restrict__ c, float* __restrict__ s,
                            float* __restrict__ V, float* __restrict__ acc,
                            unsigned* __restrict__ bar) {
  const long stride = (long)gridDim.x * blockDim.x;
  const long t0 = (long)blockIdx.x * blockDim.x + threadIdx.x;

  for (long i = t0; i < (long)G4 * KZ; i += stride) {
    int n = (int)(i / KZ), k = (int)(i % KZ);
    float v = (k < EMB + STK) ? W_ih[(long)n * (EMB + STK) + k]
                              : W_hh[(long)n * HID + (k - (EMB + STK))];
    Wc[i] = (bf16_t)v;
  }
  for (long i = t0; i < G4; i += stride) bc[i] = b_ih[i] + b_hh[i];
  for (long i = t0; i < (long)STK * HID; i += stride) Wval[i] = (bf16_t)W_val_f[i];
  for (long i = t0; i < (long)TLEN * BSZ * EMB; i += stride) {
    int t = (int)(i >> 14), b = (int)((i >> 7) & 127), e = (int)(i & 127);
    exbf[i] = (bf16_t)emb[(long)x[t * BSZ + b] * EMB + e];
  }
  for (long i = t0; i < (long)BSZ * KZ; i += stride) {
    int b = (int)(i / KZ), k = (int)(i % KZ);
    bf16_t v;
    if (k < EMB)            v = (bf16_t)emb[(long)x[b] * EMB + k];
    else if (k < EMB + STK) v = (bf16_t)1.0f;
    else                    v = (bf16_t)0.0f;
    Z[i] = v;
  }
  for (long i = t0; i < (long)BSZ * HID; i += stride) { h[i] = 0.0f; c[i] = 0.0f; }
  for (long i = t0; i < (long)BSZ * TLEN; i += stride) s[i] = 0.0f;
  for (long i = t0; i < (long)BSZ * TLEN * STK; i += stride) V[i] = 0.0f;
  for (long i = t0; i < BSZ; i += stride) acc[i] = 0.0f;
  if (t0 < 2) bar[t0] = 0u;
}

// ---------------------------------------------------------------------------
// persistent kernel: all 128 recurrent steps
//   WG wg owns Wc rows [wg*32, wg*32+32) resident in LDS for the whole run;
//   wave w computes M-tile w for the WG's two 16-wide N tiles.
// ---------------------------------------------------------------------------
__global__ void __launch_bounds__(NTHR, 1)
stack_lstm_steps(const bf16_t* __restrict__ Wc, const float* __restrict__ bc,
                 const bf16_t* __restrict__ Wval, const float* __restrict__ b_val,
                 const float* __restrict__ Wpop, const float* __restrict__ bpop,
                 const float* __restrict__ Wpush, const float* __restrict__ bpush,
                 const bf16_t* __restrict__ exbf,
                 bf16_t* __restrict__ Z, float* __restrict__ gates,
                 float* __restrict__ h, float* __restrict__ c,
                 float* __restrict__ vals, float* __restrict__ pop,
                 float* __restrict__ push, float* __restrict__ s,
                 float* __restrict__ V, unsigned* __restrict__ bar) {
  const int tid   = threadIdx.x;
  const int wg    = blockIdx.x;
  const int lane  = tid & 31;
  const int wave  = tid >> 5;        // 0..7  == M tile
  const int wglob = wg * 8 + wave;

  __shared__ bf16_t lds_bc[32 * BCP];    // Wc slab, persistent across all steps
  __shared__ bf16_t lds_bval[16 * BVP];  // Wval slab (WGs 0..7)
  __shared__ float  sc[2][2][128];
  __shared__ float  wsh[2][128];

  // ---- one-time async staging of weight slabs into LDS ----
  {
    const unsigned bc_base = (unsigned)(unsigned long long)(&lds_bc[0]);
    const char* gsrc = (const char*)(Wc + (size_t)wg * 32 * KZ);
    for (int cidx = tid; cidx < 32 * 96; cidx += NTHR) {   // 32 rows x 96 x 16B
      int row = cidx / 96, c16 = cidx % 96;
      async_copy_b128(bc_base + (unsigned)(row * (BCP * 2) + c16 * 16),
                      gsrc + (long)row * (KZ * 2) + c16 * 16);
    }
    if (wg < 8) {
      const unsigned bv_base = (unsigned)(unsigned long long)(&lds_bval[0]);
      const char* gv = (const char*)(Wval + (size_t)wg * 16 * HID);
      for (int cidx = tid; cidx < 16 * 64; cidx += NTHR) { // 16 rows x 64 x 16B
        int row = cidx / 64, c16 = cidx % 64;
        async_copy_b128(bv_base + (unsigned)(row * (BVP * 2) + c16 * 16),
                        gv + (long)row * (HID * 2) + c16 * 16);
      }
    }
    wait_async0();
    __syncthreads();
  }

  for (int t = 0; t < TLEN; ++t) {
    // ---- Phase 1: gates = Z @ Wc^T + bc  (WMMA bf16, B from LDS) ----
    {
      const int mt = wave;
      const int n0 = wg * 32;
      v8f acc0 = {}; v8f acc1 = {};
#pragma unroll 4
      for (int k0 = 0; k0 < KZ; k0 += 32) {
        v16bf a  = load_a16(Z, KZ, mt * 16, k0, lane);
        v16bf b0 = load_b16(lds_bc, BCP, 0, k0, lane);
        v16bf b1 = load_b16(lds_bc, BCP, 16, k0, lane);
        acc0 = __builtin_amdgcn_wmma_f32_16x16x32_bf16(false, a, false, b0,
                                                       (short)0, acc0, false, false);
        acc1 = __builtin_amdgcn_wmma_f32_16x16x32_bf16(false, a, false, b1,
                                                       (short)0, acc1, false, false);
      }
      const int mrow = mt * 16 + ((lane >> 4) << 3);
      const int ncol = lane & 15;
      const float bia0 = bc[n0 + ncol];
      const float bia1 = bc[n0 + 16 + ncol];
#pragma unroll
      for (int j = 0; j < 8; ++j) {
        gates[(long)(mrow + j) * G4 + n0 + ncol]      = acc0[j] + bia0;
        gates[(long)(mrow + j) * G4 + n0 + 16 + ncol] = acc1[j] + bia1;
      }
    }
    grid_barrier(bar, NWG);

    // ---- Phase 2: LSTM cell elementwise ----
    {
      for (int e = wg * NTHR + tid; e < BSZ * HID; e += NWG * NTHR) {
        const int b = e >> 9, j = e & (HID - 1);
        const float* gr = gates + (long)b * G4;
        const float gi = gr[j], gf = gr[HID + j], gg = gr[2 * HID + j], go = gr[3 * HID + j];
        const float cv = sigf(gf) * c[e] + sigf(gi) * tanhf(gg);
        const float hv = sigf(go) * tanhf(cv);
        c[e] = cv;
        h[e] = hv;
        Z[(long)b * KZ + (EMB + STK) + j] = (bf16_t)hv;
      }
    }
    grid_barrier(bar, NWG);

    // ---- Phase 3: vals = relu(h @ Wval^T + b_val) (WMMA, B from LDS); pop/push ----
    if (wg < 8) {
      const int mt = wave;
      v8f acc = {};
#pragma unroll 4
      for (int k0 = 0; k0 < HID; k0 += 32) {
        v16bf a = load_a16(Z, KZ, mt * 16, (EMB + STK) + k0, lane);
        v16bf b = load_b16(lds_bval, BVP, 0, k0, lane);
        acc = __builtin_amdgcn_wmma_f32_16x16x32_bf16(false, a, false, b,
                                                      (short)0, acc, false, false);
      }
      const int mrow = mt * 16 + ((lane >> 4) << 3);
      const int n    = wg * 16 + (lane & 15);
      const float bv = b_val[n];
#pragma unroll
      for (int j = 0; j < 8; ++j) {
        float v = acc[j] + bv;
        vals[(long)(mrow + j) * STK + n] = v > 0.0f ? v : 0.0f;
      }
    } else if (wg < 24) {
      const int b = wglob - 64;   // wg 8..23 -> b = 0..127
      float sp = 0.0f, sq = 0.0f;
      for (int k = lane; k < HID; k += 32) {
        const float hv = h[(long)b * HID + k];
        sp += hv * Wpop[k];
        sq += hv * Wpush[k];
      }
#pragma unroll
      for (int off = 16; off; off >>= 1) {
        sp += __shfl_xor(sp, off, 32);
        sq += __shfl_xor(sq, off, 32);
      }
      if (lane == 0) {
        pop[b]  = sigf(sp + bpop[0]);
        push[b] = sigf(sq + bpush[0]);
      }
    }
    grid_barrier(bar, NWG);

    // ---- Phase 4: continuous stack update + read; prepare next-step Z ----
    {
      const int row = tid >> 7;
      const int g   = tid & 127;
      const int b   = wg * 2 + row;
      const float pb = pop[b], qb = push[b];
      float sv = s[(long)b * TLEN + g];

      sc[0][row][g] = sv; __syncthreads();
      int src = 0;
#pragma unroll
      for (int off = 1; off < 128; off <<= 1) {
        float v = sc[src][row][g];
        if (g >= off) v += sc[src][row][g - off];
        sc[1 - src][row][g] = v;
        src ^= 1;
        __syncthreads();
      }
      const float tot = sc[src][row][127];
      const float suf = tot - sc[src][row][g];
      float d = pb - suf; if (d < 0.0f) d = 0.0f;
      float ns = sv - d;  if (ns < 0.0f) ns = 0.0f;
      if (g == t) ns = qb;
      V[((long)b * TLEN + t) * STK + g] = vals[(long)b * STK + g];
      __syncthreads();

      sc[0][row][g] = ns; __syncthreads();
      src = 0;
#pragma unroll
      for (int off = 1; off < 128; off <<= 1) {
        float v = sc[src][row][g];
        if (g >= off) v += sc[src][row][g - off];
        sc[1 - src][row][g] = v;
        src ^= 1;
        __syncthreads();
      }
      const float tot2 = sc[src][row][127];
      const float suf2 = tot2 - sc[src][row][g];
      float cap = 1.0f - suf2; if (cap < 0.0f) cap = 0.0f;
      const float w = ns < cap ? ns : cap;
      s[(long)b * TLEN + g] = ns;
      wsh[row][g] = w;
      __syncthreads();

      float rsum = 0.0f;
      const float* Vb = V + (long)b * TLEN * STK + g;
      for (int tp = 0; tp <= t; ++tp) rsum += wsh[row][tp] * Vb[(long)tp * STK];
      Z[(long)b * KZ + EMB + g] = (bf16_t)rsum;
      if (t + 1 < TLEN)
        Z[(long)b * KZ + g] = exbf[((long)(t + 1) * BSZ + b) * EMB + g];
      __syncthreads();
    }
    grid_barrier(bar, NWG);
  }
}

// ---------------------------------------------------------------------------
// output head: sigmoid(relu(h @ W_out^T + b_out) @ W_cls^T + b_cls)
// ---------------------------------------------------------------------------
__global__ void final_acc_kernel(const float* __restrict__ h, const float* __restrict__ W_out,
                                 const float* __restrict__ b_out, const float* __restrict__ W_cls,
                                 float* __restrict__ acc) {
  __shared__ float hsh[HID];
  __shared__ float red[256];
  const int b = blockIdx.x;
  const int tid = threadIdx.x;
  for (int k = tid; k < HID; k += 256) hsh[k] = h[(long)b * HID + k];
  __syncthreads();
  const int v = blockIdx.y * 256 + tid;
  float p = 0.0f;
  if (v < VOCAB) {
    const float* wr = W_out + (long)v * HID;
    float d = 0.0f;
#pragma unroll 8
    for (int k = 0; k < HID; ++k) d += hsh[k] * wr[k];
    d += b_out[v];
    if (d > 0.0f) p = d * W_cls[v];
  }
  red[tid] = p;
  __syncthreads();
  for (int off = 128; off; off >>= 1) {
    if (tid < off) red[tid] += red[tid + off];
    __syncthreads();
  }
  if (tid == 0) atomicAdd(&acc[b], red[0]);
}

__global__ void final_out_kernel(const float* __restrict__ acc,
                                 const float* __restrict__ b_cls,
                                 float* __restrict__ out) {
  const int b = threadIdx.x;
  if (b < BSZ) out[b] = 1.0f / (1.0f + __expf(-(acc[b] + b_cls[0])));
}

// ---------------------------------------------------------------------------
extern "C" void kernel_launch(void* const* d_in, const int* in_sizes, int n_in,
                              void* d_out, int out_size, void* d_ws, size_t ws_size,
                              hipStream_t stream) {
  (void)in_sizes; (void)n_in; (void)out_size; (void)ws_size;
  const int*   x      = (const int*)d_in[0];
  const float* emb    = (const float*)d_in[1];
  const float* W_ih   = (const float*)d_in[2];
  const float* b_ih   = (const float*)d_in[3];
  const float* W_hh   = (const float*)d_in[4];
  const float* b_hh   = (const float*)d_in[5];
  const float* W_out  = (const float*)d_in[6];
  const float* b_out  = (const float*)d_in[7];
  const float* W_push = (const float*)d_in[8];
  const float* b_push = (const float*)d_in[9];
  const float* W_pop  = (const float*)d_in[10];
  const float* b_pop  = (const float*)d_in[11];
  const float* W_val  = (const float*)d_in[12];
  const float* b_val  = (const float*)d_in[13];
  const float* W_cls  = (const float*)d_in[14];
  const float* b_cls  = (const float*)d_in[15];

  char* ws = (char*)d_ws;
  size_t off = 0;
  auto alloc = [&](size_t bytes) -> void* {
    void* p = ws + off;
    off = (off + bytes + 255) & ~(size_t)255;
    return p;
  };
  bf16_t*   Wc    = (bf16_t*)alloc((size_t)G4 * KZ * 2);
  bf16_t*   Wval  = (bf16_t*)alloc((size_t)STK * HID * 2);
  bf16_t*   exbf  = (bf16_t*)alloc((size_t)TLEN * BSZ * EMB * 2);
  bf16_t*   Z     = (bf16_t*)alloc((size_t)BSZ * KZ * 2);
  float*    gates = (float*)alloc((size_t)BSZ * G4 * 4);
  float*    h     = (float*)alloc((size_t)BSZ * HID * 4);
  float*    c     = (float*)alloc((size_t)BSZ * HID * 4);
  float*    vals  = (float*)alloc((size_t)BSZ * STK * 4);
  float*    pop   = (float*)alloc((size_t)BSZ * 4);
  float*    push  = (float*)alloc((size_t)BSZ * 4);
  float*    s     = (float*)alloc((size_t)BSZ * TLEN * 4);
  float*    V     = (float*)alloc((size_t)BSZ * TLEN * STK * 4);
  float*    bc    = (float*)alloc((size_t)G4 * 4);
  float*    acc   = (float*)alloc((size_t)BSZ * 4);
  unsigned* bar   = (unsigned*)alloc(256);

  init_kernel<<<2048, 256, 0, stream>>>(x, emb, W_ih, b_ih, W_hh, b_hh, W_val,
                                        Wc, bc, Wval, exbf, Z, h, c, s, V, acc, bar);

  stack_lstm_steps<<<NWG, NTHR, 0, stream>>>(Wc, bc, Wval, b_val, W_pop, b_pop,
                                             W_push, b_push, exbf, Z, gates, h, c,
                                             vals, pop, push, s, V, bar);

  final_acc_kernel<<<dim3(BSZ, (VOCAB + 255) / 256), 256, 0, stream>>>(h, W_out, b_out,
                                                                       W_cls, acc);
  final_out_kernel<<<1, BSZ, 0, stream>>>(acc, b_cls, (float*)d_out);
}
